// HFullDimensionMultiHeadAttention_45775761441163
// MI455X (gfx1250) — matrix operand order repaired and checked by reading, not compiled, not run on hardware
//
#include <hip/hip_runtime.h>
#include <hip/hip_bf16.h>

// ---------------------------------------------------------------------------
// Hyperbolic multi-head attention (Poincare ball) for MI455X / gfx1250.
// All GEMM stages run on v_wmma_f32_16x16x32_bf16 (bf16 A/B, f32 accum).
// Shapes fixed by the reference: B=4, S=1024, D=512, H=8.
// Round 4: sched_barrier fences keep fragment-load groups batched ahead of
// their WMMA groups (partial s_wait_loadcnt instead of full stalls).
// ---------------------------------------------------------------------------

typedef unsigned short u16;
typedef __attribute__((ext_vector_type(16))) __bf16 v16bf;
typedef __attribute__((ext_vector_type(8)))  __bf16 v8bf;
typedef __attribute__((ext_vector_type(8)))  float  v8f;

#define EPSF  1e-6f
#define MAXT  (1.0f - 1e-5f)

constexpr int Bb = 4, Ss = 1024, Dd = 512, Hh = 8;
constexpr int BS = Bb * Ss;      // 4096 rows
constexpr int HD = Hh * Dd;      // 4096

__device__ inline u16 f2bf(float x) {               // RNE f32 -> bf16
  unsigned int u = __float_as_uint(x);
  u += 0x7FFFu + ((u >> 16) & 1u);
  return (u16)(u >> 16);
}

// A-fragment (16x32 bf16, M=lane%16): two 16B chunks, 16 contiguous K values
__device__ inline v16bf load_fragA(const u16* p0, const u16* p1) {
  v8bf a = *(const v8bf*)p0;
  v8bf b = *(const v8bf*)p1;
  v16bf r;
#pragma unroll
  for (int i = 0; i < 8; ++i) { r[i] = a[i]; r[i + 8] = b[i]; }
  return r;
}

// B-fragment (32x16 bf16, N=lane%16): 16 contiguous K values (32B load)
__device__ inline v16bf load_fragB(const u16* p) { return *(const v16bf*)p; }

__device__ inline v8f wmma_bf16(v16bf a, v16bf b, v8f c) {
  return __builtin_amdgcn_wmma_f32_16x16x32_bf16(false, a, false, b,
                                                 (short)0, c, false, false);
}

// keep load groups together, ahead of the consuming WMMA group
__device__ inline void sched_fence() { __builtin_amdgcn_sched_barrier(0); }

// ---------------------------------------------------------------------------
// 1) logmap0 rows: t = atanh(min(|x|,MAXT)) * x / |x|, f32 -> bf16
// ---------------------------------------------------------------------------
__global__ __launch_bounds__(256) void logmap_rows(const float* __restrict__ x,
                                                   u16* __restrict__ t) {
  const int row  = blockIdx.x * 8 + (threadIdx.x >> 5);
  const int lane = threadIdx.x & 31;
  const float* xr = x + (size_t)row * Dd;
  float4 v[4];
  float ss = 0.f;
#pragma unroll
  for (int i = 0; i < 4; ++i) {
    v[i] = ((const float4*)xr)[lane + i * 32];
    ss += v[i].x * v[i].x + v[i].y * v[i].y + v[i].z * v[i].z + v[i].w * v[i].w;
  }
#pragma unroll
  for (int m = 16; m > 0; m >>= 1) ss += __shfl_xor(ss, m, 32);
  float n = sqrtf(fmaxf(ss, EPSF));
  float f = atanhf(fminf(n, MAXT)) / n;
  u16* tr = t + (size_t)row * Dd;
#pragma unroll
  for (int i = 0; i < 4; ++i) {
    ushort4 o;
    o.x = f2bf(v[i].x * f); o.y = f2bf(v[i].y * f);
    o.z = f2bf(v[i].z * f); o.w = f2bf(v[i].w * f);
    ((ushort4*)tr)[lane + i * 32] = o;
  }
}

// ---------------------------------------------------------------------------
// 2) weight convert + transpose: WT[b][n][k] = bf16(W[b][k][n])
// ---------------------------------------------------------------------------
__global__ void transpose_w(const float* __restrict__ W, u16* __restrict__ WT,
                            int K, int N, long total) {
  long tid = (long)blockIdx.x * 256 + threadIdx.x;
  if (tid >= total) return;
  long per = (long)K * N;
  long b = tid / per, r = tid % per;
  long n = r / K, k = r % K;
  WT[b * per + n * K + k] = f2bf(W[b * per + k * N + n]);
}

// ---------------------------------------------------------------------------
// 3) hlinear GEMM + rowwise manifold epilogue (8 waves, 16x64 per wave).
//    MODE 0: Q heads -> q_ball bf16, qn f32
//    MODE 1: K       -> k_ball bf16, kn f32
//    MODE 2: V       -> vlamT bf16 ([b][d][s], = lam*v), lamm1 f32
//    MODE 3: O       -> d_out f32 (= expmap0(lin))
// ---------------------------------------------------------------------------
template <int MODE>
__global__ __launch_bounds__(256) void hlinear_gemm(
    const u16* __restrict__ A, const u16* __restrict__ WT,
    const float* __restrict__ bias, int K,
    u16* __restrict__ outb, float* __restrict__ outf, float* __restrict__ outO) {
  const int tid = threadIdx.x, lane = tid & 31, w = tid >> 5;
  const int nl = lane & 15;
  const int hi8 = (lane & 16) ? 8 : 0;
  const int hi16 = (lane & 16) ? 16 : 0;
  const int rbase = hi8;
  const int m0 = blockIdx.x * 16;
  const int h = blockIdx.y;
  const u16* WTh = WT + (size_t)h * (size_t)K * 512;
  const float* biash = bias + (size_t)h * 512;
  const u16* Ar = A + (size_t)(m0 + nl) * K;
  const int nbase = w * 64;                 // 8 waves x 64 cols = 512

  const u16* Brow[4];
#pragma unroll
  for (int t = 0; t < 4; ++t)
    Brow[t] = WTh + (size_t)(nbase + t * 16 + nl) * K + hi16;

  v8f acc[4];
#pragma unroll
  for (int t = 0; t < 4; ++t)
#pragma unroll
    for (int i = 0; i < 8; ++i) acc[t][i] = 0.f;

  for (int k0 = 0; k0 < K; k0 += 32) {
    // batch all loads for this k-step, fence, then 4 WMMAs back-to-back
    v16bf af = load_fragA(Ar + k0 + hi16, Ar + k0 + hi16 + 8);
    v16bf bf[4];
#pragma unroll
    for (int t = 0; t < 4; ++t) bf[t] = load_fragB(Brow[t] + k0);
    sched_fence();
#pragma unroll
    for (int t = 0; t < 4; ++t) acc[t] = wmma_bf16(af, bf[t], acc[t]);
  }

  __shared__ float rowsq[16], fct[16];
  if (tid < 16) rowsq[tid] = 0.f;
  __syncthreads();

  float prt[8];
#pragma unroll
  for (int r = 0; r < 8; ++r) prt[r] = 0.f;
#pragma unroll
  for (int t = 0; t < 4; ++t) {
    float bv = biash[nbase + t * 16 + nl];
#pragma unroll
    for (int r = 0; r < 8; ++r) {
      float x = acc[t][r] + bv;
      acc[t][r] = x;
      prt[r] += x * x;
    }
  }
#pragma unroll
  for (int r = 0; r < 8; ++r) atomicAdd(&rowsq[rbase + r], prt[r]);
  __syncthreads();

  if (tid < 16) {
    float n = sqrtf(fmaxf(rowsq[tid], EPSF));
    float th = tanhf(n);
    int grow = m0 + tid;
    if (MODE == 0) {
      fct[tid] = th / n;
      int b = grow >> 10, s = grow & 1023;
      outf[((size_t)(b * Hh + h) << 10) + s] = fminf(th * th, MAXT);
    } else if (MODE == 1) {
      fct[tid] = th / n;
      outf[grow] = fminf(th * th, MAXT);
    } else if (MODE == 2) {
      float f = th / n;
      float vn = fminf(th * th, MAXT);
      float lam = 2.f / (1.f - vn);
      fct[tid] = f * lam;          // store lam * v directly
      outf[grow] = lam - 1.f;
    } else {
      fct[tid] = th / n;
    }
  }
  __syncthreads();

#pragma unroll
  for (int t = 0; t < 4; ++t) {
    int n = nbase + t * 16 + nl;
#pragma unroll
    for (int r = 0; r < 8; ++r) {
      int row = rbase + r;
      int grow = m0 + row;
      float val = acc[t][r] * fct[row];
      if (MODE == 0) {
        int b = grow >> 10, s = grow & 1023;
        outb[(((size_t)(b * Hh + h) << 10) + s) * Dd + n] = f2bf(val);
      } else if (MODE == 1) {
        outb[(size_t)grow * Dd + n] = f2bf(val);
      } else if (MODE == 2) {
        int b = grow >> 10, s = grow & 1023;
        outb[((size_t)b * Dd + n) * Ss + s] = f2bf(val);   // transposed
      } else {
        outO[(size_t)grow * Dd + n] = val;
      }
    }
  }
}

// ---------------------------------------------------------------------------
// 4) Fused flash-style hyperbolic attention per (b, h, 16-query tile).
//    8 waves split D=512 into 64-wide slices; scores reduced through LDS;
//    online max; P staged in LDS as bf16 A-fragments; PV via WMMA.
// ---------------------------------------------------------------------------
__global__ __launch_bounds__(256) void attn_kernel(
    const u16* __restrict__ qball, const float* __restrict__ qn,
    const u16* __restrict__ kball, const float* __restrict__ kn,
    const u16* __restrict__ vlamT, const float* __restrict__ lamm1,
    const float* __restrict__ stau, const float* __restrict__ sgam,
    u16* __restrict__ concat) {
  const int tid = threadIdx.x, lane = tid & 31, w = tid >> 5;
  const int nl = lane & 15;
  const int hi8 = (lane & 16) ? 8 : 0;
  const int hi16 = (lane & 16) ? 16 : 0;
  const int rbase = hi8;
  const int q0 = blockIdx.x * 16, h = blockIdx.y, b = blockIdx.z;

  __shared__ __align__(32) float ps[8][16][32];   // per-wave partial scores
  __shared__ __align__(32) u16 pS[16][32];        // softmax weights (bf16)
  __shared__ float rowm[16], scl[16], denr[16], bm[16], qnS[16], fct[16], rowsq[16];
  __shared__ float knB[32], lmB[32];

  const float itau = __expf(stau[0]);   // 1/tau = exp(scale_tau)
  const float gamma = sgam[0];

  if (tid < 16) {
    rowm[tid] = -3.0e38f;
    denr[tid] = 0.f;
    rowsq[tid] = 0.f;
    qnS[tid] = qn[((size_t)(b * Hh + h) << 10) + q0 + tid];
  }
  const u16* Qr = qball + (((size_t)(b * Hh + h) << 10) + q0 + nl) * Dd;
  const u16* Kbase = kball + (size_t)b * Ss * Dd;
  const u16* Vbase = vlamT + (size_t)b * Dd * Ss;

  // Q fragments for this wave's 64-wide D slice (key-block invariant)
  v16bf qfr[2];
#pragma unroll
  for (int dk = 0; dk < 2; ++dk) {
    int d0 = w * 64 + dk * 32;
    qfr[dk] = load_fragA(Qr + d0 + hi16, Qr + d0 + hi16 + 8);
  }

  // fixed lam*V row pointers for this wave's 4 d-tiles
  const u16* Vrow[4];
#pragma unroll
  for (int t = 0; t < 4; ++t)
    Vrow[t] = Vbase + (size_t)(w * 64 + t * 16 + nl) * Ss + hi16;

  v8f acc[4];
#pragma unroll
  for (int t = 0; t < 4; ++t)
#pragma unroll
    for (int i = 0; i < 8; ++i) acc[t][i] = 0.f;

  const int srow = tid >> 4;          // 0..15 : score row (16 threads/row)
  const int scol0 = (tid & 15) * 2;   // 2 score columns per thread
  __syncthreads();

  for (int kb = 0; kb < Ss; kb += 32) {
    if (tid < 32) {
      knB[tid] = kn[b * Ss + kb + tid];
      lmB[tid] = lamm1[b * Ss + kb + tid];
    }
    // ---- partial Q.K^T over this wave's 64-wide D slice ----
    // batch the 4 K fragments (8x b128), fence, then 4 WMMAs
    v16bf kf[4];
#pragma unroll
    for (int dk = 0; dk < 2; ++dk)
#pragma unroll
      for (int j = 0; j < 2; ++j)
        kf[dk * 2 + j] = load_fragB(Kbase + (size_t)(kb + j * 16 + nl) * Dd +
                                    w * 64 + dk * 32 + hi16);
    sched_fence();
    v8f sc[2];
#pragma unroll
    for (int j = 0; j < 2; ++j)
#pragma unroll
      for (int i = 0; i < 8; ++i) sc[j][i] = 0.f;
    sc[0] = wmma_bf16(qfr[0], kf[0], sc[0]);
    sc[1] = wmma_bf16(qfr[0], kf[1], sc[1]);
    sc[0] = wmma_bf16(qfr[1], kf[2], sc[0]);
    sc[1] = wmma_bf16(qfr[1], kf[3], sc[1]);
#pragma unroll
    for (int j = 0; j < 2; ++j)
#pragma unroll
      for (int r = 0; r < 8; ++r) ps[w][rbase + r][j * 16 + nl] = sc[j][r];

    // prefetch next key block while the VALU softmax section runs
    if (kb + 32 < Ss) {
      __builtin_prefetch(Kbase + (size_t)(kb + 32 + lane) * Dd + w * 64, 0, 3);
      __builtin_prefetch(Vrow[lane & 3] + kb + 32, 0, 3);
    }
    __syncthreads();

    // ---- reduce over 8 waves + hyperbolic distance -> similarity ----
    float sim[2];
    float lmax = -3.0e38f;
    float qv = qnS[srow];
#pragma unroll
    for (int c = 0; c < 2; ++c) {
      int col = scol0 + c;
      float dot = 0.f;
#pragma unroll
      for (int ww = 0; ww < 8; ++ww) dot += ps[ww][srow][col];
      float kv = knB[col];
      float sq = fmaxf(qv + kv - 2.f * dot, 0.f);
      float dn = fmaxf((1.f - qv) * (1.f - kv), EPSF);
      float arg = fmaxf(1.f + 2.f * sq / dn, 1.f + 1e-7f);
      sim[c] = -acoshf(arg) * itau + gamma;
      lmax = fmaxf(lmax, sim[c]);
    }
#pragma unroll
    for (int m = 1; m < 16; m <<= 1) lmax = fmaxf(lmax, __shfl_xor(lmax, m, 32));
    if ((tid & 15) == 0) bm[srow] = lmax;
    __syncthreads();

    if (tid < 16) {
      float mo = rowm[tid], mn = fmaxf(mo, bm[tid]);
      rowm[tid] = mn;
      scl[tid] = __expf(mo - mn);
    }
    __syncthreads();

    // ---- p = exp(sim - m); stage bf16 P; den update (lam-1 weighted) ----
    float mrow = rowm[srow];
    float pl = 0.f;
#pragma unroll
    for (int c = 0; c < 2; ++c) {
      float p = __expf(sim[c] - mrow);
      pS[srow][scol0 + c] = f2bf(p);
      pl += p * lmB[scol0 + c];
    }
#pragma unroll
    for (int m = 1; m < 16; m <<= 1) pl += __shfl_xor(pl, m, 32);
    if ((tid & 15) == 0) denr[srow] = denr[srow] * scl[srow] + pl;
    __syncthreads();

    // ---- rescale accum, then P @ (lam*V) for this wave's D slice ----
#pragma unroll
    for (int r = 0; r < 8; ++r) {
      float sr = scl[rbase + r];
#pragma unroll
      for (int t = 0; t < 4; ++t) acc[t][r] *= sr;
    }
    // batch the 4 lam*V fragments + P fragment, fence, then 4 WMMAs
    v16bf vf[4];
#pragma unroll
    for (int t = 0; t < 4; ++t) vf[t] = load_fragB(Vrow[t] + kb);
    v16bf pf = load_fragA(&pS[nl][hi16], &pS[nl][hi16 + 8]);
    sched_fence();
#pragma unroll
    for (int t = 0; t < 4; ++t) acc[t] = wmma_bf16(pf, vf[t], acc[t]);
    __syncthreads();
  }

  // ---- epilogue: y = num/den ; t = 0.5*atanh(min(|y|,MAXT)) * y/|y| ----
  float prt[8];
#pragma unroll
  for (int r = 0; r < 8; ++r) prt[r] = 0.f;
#pragma unroll
  for (int r = 0; r < 8; ++r) {
    float dnv = fmaxf(denr[rbase + r], EPSF);
#pragma unroll
    for (int t = 0; t < 4; ++t) {
      float y = acc[t][r] / dnv;
      acc[t][r] = y;
      prt[r] += y * y;
    }
  }
#pragma unroll
  for (int r = 0; r < 8; ++r) atomicAdd(&rowsq[rbase + r], prt[r]);
  __syncthreads();
  if (tid < 16) {
    float ny = sqrtf(fmaxf(rowsq[tid], EPSF));
    fct[tid] = 0.5f * atanhf(fminf(ny, MAXT)) / ny;
  }
  __syncthreads();
#pragma unroll
  for (int t = 0; t < 4; ++t) {
    int d = w * 64 + t * 16 + nl;
#pragma unroll
    for (int r = 0; r < 8; ++r) {
      int row = rbase + r;
      int s = q0 + row;
      float val = acc[t][r] * fct[row];
      concat[((size_t)(b * Ss + s)) * HD + h * Dd + d] = f2bf(val);
    }
  }
}

// ---------------------------------------------------------------------------
// Launch
// ---------------------------------------------------------------------------
extern "C" void kernel_launch(void* const* d_in, const int* in_sizes, int n_in,
                              void* d_out, int out_size, void* d_ws, size_t ws_size,
                              hipStream_t stream) {
  const float* query = (const float*)d_in[0];
  const float* key   = (const float*)d_in[1];
  const float* value = (const float*)d_in[2];
  const float* Wq = (const float*)d_in[3];
  const float* bq = (const float*)d_in[4];
  const float* Wk = (const float*)d_in[5];
  const float* bk = (const float*)d_in[6];
  const float* Wv = (const float*)d_in[7];
  const float* bv = (const float*)d_in[8];
  const float* Wo = (const float*)d_in[9];
  const float* bo = (const float*)d_in[10];
  const float* stau = (const float*)d_in[11];
  const float* sgam = (const float*)d_in[12];
  float* out = (float*)d_out;

  char* p = (char*)d_ws;
  auto carve = [&](size_t bytes) -> void* {
    void* r = (void*)p;
    p += (bytes + 255) & ~(size_t)255;
    return r;
  };
  u16* tq    = (u16*)carve((size_t)BS * Dd * 2);
  u16* tk    = (u16*)carve((size_t)BS * Dd * 2);
  u16* tv    = (u16*)carve((size_t)BS * Dd * 2);
  u16* WqT   = (u16*)carve((size_t)Hh * Dd * Dd * 2);
  u16* WkT   = (u16*)carve((size_t)Dd * Dd * 2);
  u16* WvT   = (u16*)carve((size_t)Dd * Dd * 2);
  u16* WoT   = (u16*)carve((size_t)HD * Dd * 2);
  u16* qballp= (u16*)carve((size_t)Bb * Hh * Ss * Dd * 2);
  float* qnp = (float*)carve((size_t)Bb * Hh * Ss * 4);
  u16* kballp= (u16*)carve((size_t)BS * Dd * 2);
  float* knp = (float*)carve((size_t)BS * 4);
  u16* vlamTp= (u16*)carve((size_t)Bb * Dd * Ss * 2);
  float* lm1p= (float*)carve((size_t)BS * 4);
  u16* cct   = (u16*)carve((size_t)BS * HD * 2);

  // 1) logmap0 of q/k/v
  logmap_rows<<<BS / 8, 256, 0, stream>>>(query, tq);
  logmap_rows<<<BS / 8, 256, 0, stream>>>(key, tk);
  logmap_rows<<<BS / 8, 256, 0, stream>>>(value, tv);

  // 2) weight transposes to bf16 [N][K]
  {
    long tWq = (long)Hh * Dd * Dd;
    transpose_w<<<(unsigned)((tWq + 255) / 256), 256, 0, stream>>>(Wq, WqT, Dd, Dd, tWq);
    long tW = (long)Dd * Dd;
    transpose_w<<<(unsigned)((tW + 255) / 256), 256, 0, stream>>>(Wk, WkT, Dd, Dd, tW);
    transpose_w<<<(unsigned)((tW + 255) / 256), 256, 0, stream>>>(Wv, WvT, Dd, Dd, tW);
    long tWo = (long)HD * Dd;
    transpose_w<<<(unsigned)((tWo + 255) / 256), 256, 0, stream>>>(Wo, WoT, HD, Dd, tWo);
  }

  // 3) projections (WMMA GEMM + expmap0 epilogues)
  hlinear_gemm<0><<<dim3(BS / 16, Hh), 256, 0, stream>>>(tq, WqT, bq, Dd, qballp, qnp, nullptr);
  hlinear_gemm<1><<<dim3(BS / 16, 1), 256, 0, stream>>>(tk, WkT, bk, Dd, kballp, knp, nullptr);
  hlinear_gemm<2><<<dim3(BS / 16, 1), 256, 0, stream>>>(tv, WvT, bv, Dd, vlamTp, lm1p, nullptr);

  // 4) fused hyperbolic flash attention -> tangent concat
  attn_kernel<<<dim3(Ss / 16, Hh, Bb), 256, 0, stream>>>(
      qballp, qnp, kballp, knp, vlamTp, lm1p, stau, sgam, cct);

  // 5) output projection: expmap0(concat_t @ Wo + bo)
  hlinear_gemm<3><<<dim3(BS / 16, 1), 256, 0, stream>>>(cct, WoT, bo, HD, nullptr, nullptr, out);
}
